// LinearAttention_42039139893667
// MI455X (gfx1250) — matrix operand order repaired and checked by reading, not compiled, not run on hardware
//
#include <hip/hip_runtime.h>
#include <math.h>

#define DIMC 64
#define HEADS 4
#define DH 32
#define HID 128
#define QKV3 384
#define NSP 131072
#define EPSV 1e-5f

typedef __attribute__((ext_vector_type(16))) __bf16 v16bf;
typedef __attribute__((ext_vector_type(8)))  __bf16 v8bf;
typedef __attribute__((ext_vector_type(8)))  float  v8f;
typedef __attribute__((ext_vector_type(4)))  float  v4f;
typedef __attribute__((ext_vector_type(4)))  unsigned int u32x4;
typedef __attribute__((ext_vector_type(8)))  int i32x8;
typedef __attribute__((ext_vector_type(4)))  int i32x4;

static __device__ __forceinline__ v16bf frag_from(const __bf16* p0, const __bf16* p1) {
  v8bf a = *(const v8bf*)p0;
  v8bf b = *(const v8bf*)p1;
  return __builtin_shufflevector(a, b, 0,1,2,3,4,5,6,7,8,9,10,11,12,13,14,15);
}

static __device__ __forceinline__ v16bf frag_from_f32(const float* p0, const float* p1) {
  v4f a0 = *(const v4f*)p0, a1 = *(const v4f*)(p0 + 4);
  v4f b0 = *(const v4f*)p1, b1 = *(const v4f*)(p1 + 4);
  v16bf r;
#pragma unroll
  for (int i = 0; i < 4; ++i) {
    r[i]      = (__bf16)a0[i];
    r[i + 4]  = (__bf16)a1[i];
    r[i + 8]  = (__bf16)b0[i];
    r[i + 12] = (__bf16)b1[i];
  }
  return r;
}

static __device__ __forceinline__ v8f wmma_bf16(v16bf a, v16bf b, v8f c) {
  return __builtin_amdgcn_wmma_f32_16x16x32_bf16(false, a, false, b, (short)0, c, false, false);
}

// ---------------------------------------------------------------------------
// Kernel 1: LayerNorm(channel) + QKV projection (384x64 @ 64xN) -> bf16 qkv ws
// block = 256 threads (8 waves), 128 columns per block, 16 columns per wave.
// x tile (64 rows x 128 cols fp32, row stride NSP) staged to LDS via the
// Tensor Data Mover with LDS padding 128 DW + 4 DW -> 132-float rows.
// ---------------------------------------------------------------------------
__global__ __launch_bounds__(256) void k_ln_qkv(const float* __restrict__ x,
    const float* __restrict__ g1, const float* __restrict__ b1v,
    const float* __restrict__ wqkv, __bf16* __restrict__ qkv)
{
  __shared__ float xt[DIMC * 132];
  __shared__ __bf16 bt[128 * 72];
  __shared__ float sg[DIMC], sb[DIMC];

  const int tid  = threadIdx.x;
  const int lane = tid & 31;
  const int wv   = tid >> 5;
  const int blocksPerB = NSP / 128;
  const int b  = blockIdx.x / blocksPerB;
  const int n0 = (blockIdx.x % blocksPerB) * 128;

  if (tid < DIMC) { sg[tid] = g1[tid]; sb[tid] = b1v[tid]; }
  const float* xb = x + (size_t)b * DIMC * NSP + n0;

#if defined(__HIP_DEVICE_COMPILE__) && __has_builtin(__builtin_amdgcn_tensor_load_to_lds)
  if (wv == 0) {
    // Tensor DMA descriptor (D#): 2D tile 128x64 of 4B elements, row stride NSP,
    // LDS padding: 4 DWORDs after every 128 DWORDs (-> xt row stride 132 floats).
    unsigned long long ga = (unsigned long long)(uintptr_t)xb;
    unsigned int ldsa = (unsigned int)(uintptr_t)&xt[0];  // flat LDS addr low 32 = LDS offset
    u32x4 g0;
    g0[0] = 1u;                                           // count=1, user descriptor
    g0[1] = ldsa;                                         // lds_addr
    g0[2] = (unsigned int)(ga & 0xffffffffu);             // global_addr[31:0]
    g0[3] = (unsigned int)((ga >> 32) & 0x01ffffffu) | (2u << 30); // addr[56:32], type=2
    i32x8 g1d;
    g1d[0] = (int)((2u << 16) | (1u << 20) | (6u << 22) | (3u << 25)); // 4B, pad 128DW+4DW
    g1d[1] = (int)((131072u & 0xffffu) << 16);            // tensor_dim0[15:0]
    g1d[2] = (int)((131072u >> 16) | (64u << 16));        // tensor_dim0[31:16], tensor_dim1 lo
    g1d[3] = (int)(128u << 16);                           // tile_dim0 = 128
    g1d[4] = (int)64u;                                    // tile_dim1 = 64, tile_dim2 = 0
    g1d[5] = (int)131072u;                                // tensor_dim0_stride[31:0] = NSP
    g1d[6] = 0;
    g1d[7] = 0;
    i32x4 gz = {};
#if __clang_major__ >= 23
    i32x8 gz8 = {};
    __builtin_amdgcn_tensor_load_to_lds(g0, g1d, gz, gz, gz8, 0);
#else
    __builtin_amdgcn_tensor_load_to_lds(g0, g1d, gz, gz, 0);
#endif
    __builtin_amdgcn_s_wait_tensorcnt(0);
  }
#else
  for (int i = tid; i < DIMC * 128; i += 256) {
    int c = i >> 7, j2 = i & 127;
    xt[c * 132 + j2] = xb[(size_t)c * NSP + j2];
  }
#endif
  __syncthreads();

  // per-column LayerNorm; lane pair (L, L^16) covers the 64 channels of col j
  const int j = wv * 16 + (lane & 15);
  const int chalf = (lane < 16) ? 0 : 32;
  float s1 = 0.f, s2 = 0.f;
  float vals[32];
#pragma unroll
  for (int i = 0; i < 32; ++i) {
    float v = xt[(chalf + i) * 132 + j];
    vals[i] = v; s1 += v; s2 += v * v;
  }
  s1 += __shfl_xor(s1, 16, 32);
  s2 += __shfl_xor(s2, 16, 32);
  float mean = s1 * (1.f / 64.f);
  float var  = s2 * (1.f / 64.f) - mean * mean;
  float rstd = rsqrtf(var + EPSV);
#pragma unroll
  for (int i8 = 0; i8 < 4; ++i8) {
    v8bf pk;
#pragma unroll
    for (int t = 0; t < 8; ++t) {
      int c = chalf + i8 * 8 + t;
      pk[t] = (__bf16)((vals[i8 * 8 + t] - mean) * rstd * sg[c] + sb[c]);
    }
    *(v8bf*)&bt[j * 72 + chalf + i8 * 8] = pk;  // same-wave DS ops stay in order
  }

  // QKV GEMM: 24 M-tiles of 16, K=64 (2 wmma k-steps of 32)
  const int ka = (lane < 16) ? 0 : 8;
  const int kb = (lane < 16) ? 0 : 16;
  const int m  = lane & 15;
  const int rbase = (lane < 16) ? 0 : 8;
  __bf16* qout = qkv + (size_t)b * QKV3 * NSP + n0;
  for (int mt = 0; mt < 24; ++mt) {
    v8f acc = {};
    const float* wrow = wqkv + (mt * 16 + m) * 64;
#pragma unroll
    for (int ks = 0; ks < 2; ++ks) {
      v16bf a  = frag_from_f32(wrow + ks * 32 + ka, wrow + ks * 32 + ka + 16);
      v16bf bb = frag_from(&bt[j * 72 + ks * 32 + ka], &bt[j * 72 + ks * 32 + ka + 16]);
      acc = wmma_bf16(a, bb, acc);
    }
#pragma unroll
    for (int r = 0; r < 8; ++r) {
      int o = mt * 16 + rbase + r;
      qout[(size_t)o * NSP + j] = (__bf16)acc[r];
    }
  }
}

// ---------------------------------------------------------------------------
// Kernel 2: per-(b,h,d) row max & sum(exp) over n (k softmax stats), 256 rows
// 16B vector loads: 256 thr x 8 bf16 = 2048 elems/iter, 64 iters per pass.
// ---------------------------------------------------------------------------
__global__ __launch_bounds__(256) void k_kstats(const __bf16* __restrict__ qkv,
    float* __restrict__ kmax, float* __restrict__ ksum)
{
  __shared__ float red[256];
  const int r = blockIdx.x;             // r = b*128 + h*32 + d
  const int b = r >> 7;
  const int o = HID + (r & 127);
  const __bf16* row = qkv + ((size_t)b * QKV3 + o) * NSP;
  const int tid = threadIdx.x;
  float m = -3.4e38f;
  for (int i = tid * 8; i < NSP; i += 256 * 8) {
    v8bf v = *(const v8bf*)(row + i);
#pragma unroll
    for (int t = 0; t < 8; ++t) m = fmaxf(m, (float)v[t]);
  }
  red[tid] = m; __syncthreads();
  for (int s = 128; s > 0; s >>= 1) {
    if (tid < s) red[tid] = fmaxf(red[tid], red[tid + s]);
    __syncthreads();
  }
  m = red[0]; __syncthreads();
  float s = 0.f;
  for (int i = tid * 8; i < NSP; i += 256 * 8) {
    v8bf v = *(const v8bf*)(row + i);
#pragma unroll
    for (int t = 0; t < 8; ++t) s += __expf((float)v[t] - m);
  }
  red[tid] = s; __syncthreads();
  for (int st = 128; st > 0; st >>= 1) {
    if (tid < st) red[tid] += red[tid + st];
    __syncthreads();
  }
  if (tid == 0) { kmax[r] = m; ksum[r] = red[0]; }
}

// ---------------------------------------------------------------------------
// Kernel 3: ctx partials: ctx[d][e] = sum_n exp(k[d,n]-max_d) * v[e,n]
// WMMA over the n dimension (K). Per-wave 32x32 partial -> ws (deterministic).
// grid = 8 (b,h) * 128 chunks; each wave handles 128 n (4 k-steps of 32).
// ---------------------------------------------------------------------------
__global__ __launch_bounds__(256) void k_ctx(const __bf16* __restrict__ qkv,
    const float* __restrict__ kmax, float* __restrict__ part)
{
  const int SPLIT = 128;
  const int bh    = blockIdx.x / SPLIT;
  const int chunk = blockIdx.x % SPLIT;
  const int b = bh >> 2, h = bh & 3;
  const int lane = threadIdx.x & 31, wv = threadIdx.x >> 5;
  const int nbase = chunk * 1024 + wv * 128;
  const int ka = (lane < 16) ? 0 : 8;
  const int kb = (lane < 16) ? 0 : 16;
  const int dm = lane & 15;

  const __bf16* krow0 = qkv + ((size_t)b * QKV3 + HID + h * 32) * NSP;
  const __bf16* vrow0 = qkv + ((size_t)b * QKV3 + 2 * HID + h * 32) * NSP;
  float mx[2];
  mx[0] = kmax[b * 128 + h * 32 + dm];
  mx[1] = kmax[b * 128 + h * 32 + dm + 16];

  v8f acc[2][2] = {};
  for (int ks = 0; ks < 4; ++ks) {
    const int nb = nbase + ks * 32;
    v16bf afr[2], bfr[2];
#pragma unroll
    for (int dt = 0; dt < 2; ++dt) {
      const __bf16* kp = krow0 + (size_t)(dm + 16 * dt) * NSP + nb;
      v16bf raw = frag_from(kp + ka, kp + ka + 16);
      v16bf e;
#pragma unroll
      for (int i = 0; i < 16; ++i) e[i] = (__bf16)__expf((float)raw[i] - mx[dt]);
      afr[dt] = e;
    }
#pragma unroll
    for (int et = 0; et < 2; ++et) {
      const __bf16* vp = vrow0 + (size_t)((lane & 15) + 16 * et) * NSP + nb;
      bfr[et] = frag_from(vp + ka, vp + ka + 16);
    }
#pragma unroll
    for (int dt = 0; dt < 2; ++dt)
#pragma unroll
      for (int et = 0; et < 2; ++et)
        acc[dt][et] = wmma_bf16(afr[dt], bfr[et], acc[dt][et]);
  }
  const int rbase = (lane < 16) ? 0 : 8;
  float* pbase = part + ((size_t)bh * 1024 + chunk * 8 + wv) * 1024;
#pragma unroll
  for (int dt = 0; dt < 2; ++dt)
#pragma unroll
    for (int et = 0; et < 2; ++et)
#pragma unroll
      for (int r = 0; r < 8; ++r) {
        int d = dt * 16 + rbase + r;
        int e = et * 16 + (lane & 15);
        pbase[d * 32 + e] = acc[dt][et][r];
      }
}

// ---------------------------------------------------------------------------
// Kernel 4: deterministic reduction of the 1024 ctx partials per (b,h)
// ---------------------------------------------------------------------------
__global__ __launch_bounds__(256) void k_ctxred(const float* __restrict__ part,
                                                float* __restrict__ ctx)
{
  int i = blockIdx.x * 256 + threadIdx.x;      // 0..8191 : (bh, d*32+e)
  if (i >= 8 * 1024) return;
  int bh = i >> 10, de = i & 1023;
  float s = 0.f;
  for (int c = 0; c < 1024; ++c) s += part[((size_t)bh * 1024 + c) * 1024 + de];
  ctx[i] = s;
}

// ---------------------------------------------------------------------------
// Kernel 5: q softmax(d) * scale, out = ctx^T q, y = w_out @ out + b, LN2
// ---------------------------------------------------------------------------
__global__ __launch_bounds__(256) void k_out(const __bf16* __restrict__ qkv,
    const float* __restrict__ ctx, const float* __restrict__ ksum,
    const float* __restrict__ wout, const float* __restrict__ bout,
    const float* __restrict__ g2, const float* __restrict__ b2v,
    float* __restrict__ y)
{
  __shared__ __bf16 ctxA[HEADS * 32 * 40];   // [h][e][d], scaled by 1/sum_d
  __shared__ __bf16 outBt[128 * 136];        // [col][o] for second GEMM B
  __shared__ float sbo[DIMC], sg[DIMC], sb[DIMC];

  const int tid  = threadIdx.x;
  const int lane = tid & 31, wv = tid >> 5;
  const int blocksPerB = NSP / 128;
  const int b  = blockIdx.x / blocksPerB;
  const int n0 = (blockIdx.x % blocksPerB) * 128;

  if (tid < DIMC) { sbo[tid] = bout[tid]; sg[tid] = g2[tid]; sb[tid] = b2v[tid]; }
  for (int i = tid; i < HEADS * 32 * 32; i += 256) {
    int h = i >> 10, rem = i & 1023;
    int d = rem >> 5, e = rem & 31;
    float v = ctx[((b * 4 + h) * 32 + d) * 32 + e] / ksum[b * 128 + h * 32 + d];
    ctxA[(h * 32 + e) * 40 + d] = (__bf16)v;
  }
  __syncthreads();

  const int ka = (lane < 16) ? 0 : 8;
  const int kb = (lane < 16) ? 0 : 16;
  const int jl = wv * 16 + (lane & 15);
  const int rbase = (lane < 16) ? 0 : 8;
  const float scale = 0.17677669529663687f;  // 32^-0.5

  const __bf16* qb = qkv + (size_t)b * QKV3 * NSP + n0 + jl;
  for (int h = 0; h < HEADS; ++h) {
    // softmax over d (32) for column jl: lane pair (L, L^16) holds complementary halves
    float f[16];
#pragma unroll
    for (int i = 0; i < 16; ++i)
      f[i] = (float)qb[(size_t)(h * 32 + kb + i) * NSP];
    float m = f[0];
#pragma unroll
    for (int i = 1; i < 16; ++i) m = fmaxf(m, f[i]);
    m = fmaxf(m, __shfl_xor(m, 16, 32));
    float s = 0.f;
#pragma unroll
    for (int i = 0; i < 16; ++i) { f[i] = __expf(f[i] - m); s += f[i]; }
    s += __shfl_xor(s, 16, 32);
    float inv = scale / s;
    v16bf bq;
#pragma unroll
    for (int i = 0; i < 16; ++i) bq[i] = (__bf16)(f[i] * inv);

#pragma unroll
    for (int et = 0; et < 2; ++et) {
      const __bf16* ap = &ctxA[(h * 32 + 16 * et + (lane & 15)) * 40];
      v16bf a = frag_from(ap + kb, ap + kb + 8);
      v8f acc = {};
      acc = wmma_bf16(a, bq, acc);
      v8bf pk;
#pragma unroll
      for (int r = 0; r < 8; ++r) pk[r] = (__bf16)acc[r];
      *(v8bf*)&outBt[jl * 136 + h * 32 + et * 16 + rbase] = pk;
    }
  }

  // y = w_out(64x128) @ out + b_out ; then LayerNorm over 64 channels
  float yv[4][8];
#pragma unroll
  for (int mt = 0; mt < 4; ++mt) {
    v8f acc = {};
    const float* wrow = wout + (mt * 16 + (lane & 15)) * HID;
#pragma unroll
    for (int ks = 0; ks < 4; ++ks) {
      v16bf a  = frag_from_f32(wrow + ks * 32 + ka, wrow + ks * 32 + ka + 16);
      v16bf bb = frag_from(&outBt[jl * 136 + ks * 32 + ka],
                           &outBt[jl * 136 + ks * 32 + ka + 16]);
      acc = wmma_bf16(a, bb, acc);
    }
#pragma unroll
    for (int r = 0; r < 8; ++r) yv[mt][r] = acc[r] + sbo[mt * 16 + rbase + r];
  }
  float s1 = 0.f, s2 = 0.f;
#pragma unroll
  for (int mt = 0; mt < 4; ++mt)
#pragma unroll
    for (int r = 0; r < 8; ++r) { float v = yv[mt][r]; s1 += v; s2 += v * v; }
  s1 += __shfl_xor(s1, 16, 32);
  s2 += __shfl_xor(s2, 16, 32);
  float mean = s1 * (1.f / 64.f);
  float var  = s2 * (1.f / 64.f) - mean * mean;
  float rstd = rsqrtf(var + EPSV);
  float* yb = y + (size_t)b * DIMC * NSP + n0 + jl;
#pragma unroll
  for (int mt = 0; mt < 4; ++mt)
#pragma unroll
    for (int r = 0; r < 8; ++r) {
      int c = mt * 16 + rbase + r;
      yb[(size_t)c * NSP] = (yv[mt][r] - mean) * rstd * sg[c] + sb[c];
    }
}

// ---------------------------------------------------------------------------
extern "C" void kernel_launch(void* const* d_in, const int* in_sizes, int n_in,
                              void* d_out, int out_size, void* d_ws, size_t ws_size,
                              hipStream_t stream)
{
  (void)in_sizes; (void)n_in; (void)out_size; (void)ws_size;
  const float* x    = (const float*)d_in[0];
  const float* g1   = (const float*)d_in[1];
  const float* b1v  = (const float*)d_in[2];
  const float* wqkv = (const float*)d_in[3];
  const float* wout = (const float*)d_in[4];
  const float* bout = (const float*)d_in[5];
  const float* g2   = (const float*)d_in[6];
  const float* b2v  = (const float*)d_in[7];
  float* y = (float*)d_out;

  char* ws = (char*)d_ws;
  __bf16* qkv = (__bf16*)ws;                       // 2*384*131072 bf16 = 192 MiB
  size_t off = (size_t)2 * QKV3 * NSP * sizeof(__bf16);
  float* part = (float*)(ws + off); off += (size_t)8 * 1024 * 1024 * sizeof(float); // 32 MiB
  float* kmax = (float*)(ws + off); off += 256 * sizeof(float);
  float* ksum = (float*)(ws + off); off += 256 * sizeof(float);
  float* ctx  = (float*)(ws + off); off += 8 * 32 * 32 * sizeof(float);

  const int blocksPerB = NSP / 128;
  k_ln_qkv <<<2 * blocksPerB, 256, 0, stream>>>(x, g1, b1v, wqkv, qkv);
  k_kstats <<<256,            256, 0, stream>>>(qkv, kmax, ksum);
  k_ctx    <<<8 * 128,        256, 0, stream>>>(qkv, kmax, part);
  k_ctxred <<<32,             256, 0, stream>>>(part, ctx);
  k_out    <<<2 * blocksPerB, 256, 0, stream>>>(qkv, ctx, ksum, wout, bout, g2, b2v, y);
}
